// HRNet_W48_Proto_42460046688991
// MI455X (gfx1250) — matrix operand order repaired and checked by reading, not compiled, not run on hardware
//
#include <hip/hip_runtime.h>
#include <hip/hip_bf16.h>

// ---------------- problem constants ----------------
#define NPIX   131072      // 4*128*256
#define DIMD   720
#define DIMP   736         // 720 padded to 23*32 for WMMA k-loop
#define NCLS   19
#define NPROT  10
#define NJ     190         // NCLS*NPROT
#define HW     32768       // 128*256
#define BROW   40          // LDS B-tile row stride in halfs (80B: bank-conflict-free, 16B aligned)

// CDNA5 async global->LDS staging (ASYNCcnt) if the toolchain exposes it.
#if __has_builtin(__builtin_amdgcn_global_load_async_to_lds_b128) && \
    __has_builtin(__builtin_amdgcn_s_wait_asynccnt)
#define HAVE_ASYNC_LDS 1
#else
#define HAVE_ASYNC_LDS 0
#endif

typedef __attribute__((ext_vector_type(8)))  unsigned short v8u;
typedef __attribute__((ext_vector_type(16))) unsigned short v16u;
typedef __attribute__((ext_vector_type(16))) __bf16        v16bf;
typedef __attribute__((ext_vector_type(8)))  float         v8f;

// ---------------- helpers ----------------
__device__ __forceinline__ unsigned short f2bf(float f) {
    unsigned u = __builtin_bit_cast(unsigned, f);
    u += 0x7FFFu + ((u >> 16) & 1u);                // round to nearest even
    return (unsigned short)(u >> 16);
}
__device__ __forceinline__ float bf2f(unsigned short h) {
    unsigned u = ((unsigned)h) << 16;
    return __builtin_bit_cast(float, u);
}
__device__ __forceinline__ float waveRedSum(float v) {
    for (int o = 16; o > 0; o >>= 1) v += __shfl_down(v, o, 32);
    return v;
}
// 256-thread block sum; lds must hold >= 8 floats
__device__ __forceinline__ float blockRedSum256(float v, float* lds) {
    const int lane = threadIdx.x & 31, wv = threadIdx.x >> 5;
    v = waveRedSum(v);
    __syncthreads();
    if (lane == 0) lds[wv] = v;
    __syncthreads();
    if (threadIdx.x == 0) {
        float s = 0.f;
        for (int i = 0; i < 8; i++) s += lds[i];
        lds[0] = s;
    }
    __syncthreads();
    return lds[0];
}

#if HAVE_ASYNC_LDS
typedef int vi4 __attribute__((vector_size(16)));    // matches builtin param pointee
__device__ __forceinline__ void stageAsync16B(const unsigned short* g, unsigned short* l) {
    __builtin_amdgcn_global_load_async_to_lds_b128(
        (__attribute__((address_space(1))) vi4*)g,
        (__attribute__((address_space(3))) vi4*)l, 0, 0);
}
#endif

// B fragment (32x16 bf16) from LDS tile: lane l -> col l&15, K = {koff..koff+7, 16+koff..16+koff+7}
__device__ __forceinline__ v16bf ldsFragB(const unsigned short* bTile, int t, int lane, int koff) {
    const unsigned short* bp = &bTile[(t * 16 + (lane & 15)) * BROW + koff];
    const v8u blo = *(const v8u*)(bp);
    const v8u bhi = *(const v8u*)(bp + 16);
    const v16u bu = __builtin_shufflevector(blo, bhi,
                    0,1,2,3,4,5,6,7,8,9,10,11,12,13,14,15);
    return __builtin_bit_cast(v16bf, bu);
}

// ---------------- K1: layernorm + l2norm of features -> bf16 padded rows ----------------
// grid 4096 x 256 threads; block handles 32 consecutive pixels (same b,h).
__global__ void featNormKernel(const float* __restrict__ c,
                               const float* __restrict__ g,
                               const float* __restrict__ be,
                               unsigned short* __restrict__ wsC) {
    const int lane = threadIdx.x & 31, wv = threadIdx.x >> 5;
    const int n0 = blockIdx.x * 32;
    const int b = n0 >> 15, rem = n0 & 32767, h = rem >> 8, w0 = rem & 255;

    __shared__ unsigned short tile[32][730];   // pad 730: stride 1460B coprime w/ banks
    __shared__ float rsum[8][32], rsq[8][32];
    __shared__ float mu[32], rs[32], linv[32];

    // load (coalesced) + running sums; wave wv reads d = wv + 8*j, lane -> pixel
    float s = 0.f, sq = 0.f;
    const size_t baseIdx = (((size_t)b * DIMD) * 128 + h) * 256 + w0 + lane;
    for (int j = 0; j < 90; j++) {
        const int d = wv + 8 * j;
        const float v = c[baseIdx + (size_t)d * HW];
        tile[lane][d] = f2bf(v);
        s += v; sq += v * v;
    }
    rsum[wv][lane] = s; rsq[wv][lane] = sq;
    __syncthreads();
    if (threadIdx.x < 32) {
        const int p = threadIdx.x;
        float ts = 0.f, tq = 0.f;
        for (int i = 0; i < 8; i++) { ts += rsum[i][p]; tq += rsq[i][p]; }
        const float m = ts * (1.f / 720.f);
        const float var = tq * (1.f / 720.f) - m * m;
        mu[p] = m; rs[p] = rsqrtf(var + 1e-5f);
    }
    __syncthreads();

    // normalize: wave wv owns d in [wv*90, wv*90+90), pixel = lane (conflict-free)
    const float m_ = mu[lane], r_ = rs[lane];
    float l2 = 0.f;
    for (int j = 0; j < 90; j++) {
        const int d = wv * 90 + j;
        const float x = bf2f(tile[lane][d]);
        const float y = (x - m_) * r_ * g[d] + be[d];
        tile[lane][d] = f2bf(y);
        l2 += y * y;
    }
    rsq[wv][lane] = l2;
    __syncthreads();
    if (threadIdx.x < 32) {
        const int p = threadIdx.x;
        float t = 0.f;
        for (int i = 0; i < 8; i++) t += rsq[i][p];
        linv[p] = 1.f / fmaxf(sqrtf(t), 1e-12f);
    }
    __syncthreads();

    // write padded bf16 rows (coalesced): wave wv writes pixels wv, wv+8, wv+16, wv+24
    for (int pi = 0; pi < 4; pi++) {
        const int p = wv + 8 * pi;
        const float li = linv[p];
        const size_t rowOff = (size_t)(n0 + p) * DIMP;
        for (int d = lane; d < DIMP; d += 32) {
            const float y = (d < DIMD) ? bf2f(tile[p][d]) * li : 0.f;
            wsC[rowOff + d] = f2bf(y);
        }
    }
}

// ---------------- K2: l2norm prototypes -> bf16 padded 192 x 736 ----------------
__global__ void protoNormKernel(const float* __restrict__ protos,
                                unsigned short* __restrict__ wsP) {
    __shared__ float red[8];
    const int j = blockIdx.x;                  // 0..191
    if (j >= NJ) {
        for (int d = threadIdx.x; d < DIMP; d += 256) wsP[(size_t)j * DIMP + d] = 0;
        return;
    }
    const float* row = protos + (size_t)j * DIMD;
    float ss = 0.f;
    for (int d = threadIdx.x; d < DIMD; d += 256) { const float v = row[d]; ss += v * v; }
    ss = blockRedSum256(ss, red);
    const float rinv = 1.f / fmaxf(sqrtf(ss), 1e-12f);
    for (int d = threadIdx.x; d < DIMP; d += 256)
        wsP[(size_t)j * DIMP + d] = (d < DIMD) ? f2bf(row[d] * rinv) : (unsigned short)0;
}

// ---------------- K3: WMMA GEMM proto_logits = _c (Nx736 bf16) x protosT (736x192 bf16) ----
// grid N/128 blocks x 256 threads. Each block: 128 rows x all 192 cols.
// B slice (192x32 bf16) staged per k-step in LDS (async double-buffered when available);
// each wave keeps one A fragment in registers, reused across 12 column tiles.
__global__ void __launch_bounds__(256)
gemmKernel(const unsigned short* __restrict__ wsC,
           const unsigned short* __restrict__ wsP,
           float* __restrict__ outPL) {
    const int lane = threadIdx.x & 31, wv = threadIdx.x >> 5;
    const int rowBase = blockIdx.x * 128 + wv * 16;

#if HAVE_ASYNC_LDS
    __shared__ unsigned short bTile[2][192 * BROW];   // ping-pong, 2 x 15,360 B
#else
    __shared__ unsigned short bTile[1][192 * BROW];
#endif

    // A fragment (16x32 bf16): lane l holds row l&15; K = {ab..ab+7, 16+ab..16+ab+7}
    const int abase  = (lane < 16) ? 0 : 8;
    const int koff   = (lane < 16) ? 0 : 8;
    const unsigned short* aPtr = wsC + (size_t)(rowBase + (lane & 15)) * DIMP;

    v8f acc[12];
#pragma unroll
    for (int t = 0; t < 12; t++) acc[t] = (v8f){};

#if HAVE_ASYNC_LDS
    // prologue: async-stage slice kk=0 into buffer 0
#pragma unroll
    for (int i = 0; i < 3; i++) {
        const int chunk = threadIdx.x + i * 256;        // 0..767 (16B chunks)
        const int row = chunk >> 2, part = chunk & 3;
        stageAsync16B(wsP + (size_t)row * DIMP + part * 8,
                      &bTile[0][row * BROW + part * 8]);
    }
    __builtin_amdgcn_s_wait_asynccnt(0);
    __syncthreads();
#endif

    int s = 0;
    for (int kk = 0; kk < DIMP; kk += 32, s++) {
#if HAVE_ASYNC_LDS
        const unsigned short* bCur = bTile[s & 1];
        if (kk + 32 < DIMP) {
            unsigned short* bNxt = bTile[(s + 1) & 1];
#pragma unroll
            for (int i = 0; i < 3; i++) {
                const int chunk = threadIdx.x + i * 256;
                const int row = chunk >> 2, part = chunk & 3;
                stageAsync16B(wsP + (size_t)row * DIMP + (kk + 32) + part * 8,
                              &bNxt[row * BROW + part * 8]);
            }
        }
#else
        const unsigned short* bCur = bTile[0];
        __syncthreads();
#pragma unroll
        for (int i = 0; i < 3; i++) {
            const int chunk = threadIdx.x + i * 256;        // 0..767 (16B chunks)
            const int row = chunk >> 2, part = chunk & 3;
            const v8u v = *(const v8u*)(wsP + (size_t)row * DIMP + kk + part * 8);
            *(v8u*)(&bTile[0][row * BROW + part * 8]) = v;
        }
        __syncthreads();
#endif

        // ---- A fragment (reused for all 12 column tiles) ----
        __builtin_prefetch((const void*)(aPtr + kk + 32), 0, 1);  // next K slice of A
        const v8u alo = *(const v8u*)(aPtr + kk + abase);
        const v8u ahi = *(const v8u*)(aPtr + kk + 16 + abase);
        const v16u au = __builtin_shufflevector(alo, ahi,
                        0,1,2,3,4,5,6,7,8,9,10,11,12,13,14,15);
        const v16bf a = __builtin_bit_cast(v16bf, au);

        // ---- 2-deep pipelined B fragments + 12 WMMAs ----
        v16bf b0 = ldsFragB(bCur, 0, lane, koff);
#pragma unroll
        for (int t = 0; t < 12; t++) {
            const v16bf b1 = (t < 11) ? ldsFragB(bCur, t + 1, lane, koff) : b0;
            acc[t] = __builtin_amdgcn_wmma_f32_16x16x32_bf16(false, a, false, b0,
                                                             (short)0, acc[t], false, false);
            b0 = b1;
        }

#if HAVE_ASYNC_LDS
        __builtin_amdgcn_s_wait_asynccnt(0);
        __syncthreads();
#endif
    }

    // C/D layout: VGPR i -> row i (lanes 0-15) / i+8 (lanes 16-31); col = lane&15
    const int mOff = (lane < 16) ? 0 : 8;
    const int colIn = lane & 15;
#pragma unroll
    for (int t = 0; t < 12; t++) {
        const int col = t * 16 + colIn;
        if (col < NJ) {
            for (int i = 0; i < 8; i++)
                outPL[(size_t)(rowBase + mOff + i) * NJ + col] = acc[t][i];
        }
    }
}

// ---------------- K4: class-max + layernorm(19) + argmax + class lists ----------------
// grid N/8 x 256; one wave per pixel.
__global__ void segPostKernel(const float* __restrict__ pl,
                              const int* __restrict__ gt,
                              const float* __restrict__ mg,
                              const float* __restrict__ mb,
                              float* __restrict__ outSeg,
                              int* __restrict__ correct,
                              int* __restrict__ counts,
                              int* __restrict__ lists) {
    const int lane = threadIdx.x & 31, wv = threadIdx.x >> 5;
    const int n = blockIdx.x * 8 + wv;
    __shared__ float sv[8][19];
    __shared__ float smu[8], srs[8];
    __shared__ int   sarg[8];

    const float* row = pl + (size_t)n * NJ;
    if (lane < NCLS) {
        float mx = row[lane * NPROT];
        for (int i = 1; i < NPROT; i++) mx = fmaxf(mx, row[lane * NPROT + i]);
        sv[wv][lane] = mx;
    }
    __syncthreads();
    if (lane == 0) {
        float s = 0.f, q = 0.f;
        for (int k = 0; k < NCLS; k++) { const float v = sv[wv][k]; s += v; q += v * v; }
        const float m = s / 19.f, var = q / 19.f - m * m, r = rsqrtf(var + 1e-5f);
        smu[wv] = m; srs[wv] = r;
        int bi = 0; float bv = -1e30f;
        for (int k = 0; k < NCLS; k++) {
            const float y = (sv[wv][k] - m) * r * mg[k] + mb[k];
            if (y > bv) { bv = y; bi = k; }
        }
        sarg[wv] = bi;
    }
    __syncthreads();
    const int b = n >> 15, hw = n & 32767;
    if (lane < NCLS) {
        const float y = (sv[wv][lane] - smu[wv]) * srs[wv] * mg[lane] + mb[lane];
        outSeg[((size_t)b * NCLS + lane) * HW + hw] = y;
    }
    if (lane == 0) {
        const int g = gt[n];
        correct[n] = (g == sarg[wv]) ? 1 : 0;
        const int pos = atomicAdd(&counts[g], 1);
        lists[(size_t)g * NPIX + pos] = n;
    }
}

// ---------------- K5: per-class masked sinkhorn + argmax -> picked, proto_target ----------------
__global__ void __launch_bounds__(1024)
sinkhornKernel(const float* __restrict__ pl,
               const int* __restrict__ counts,
               const int* __restrict__ lists,
               float* __restrict__ bbuf,
               int* __restrict__ picked,
               float* __restrict__ outPT) {
    const int k = blockIdx.x;
    const int tid = threadIdx.x, lane = tid & 31, wv = tid >> 5;
    __shared__ float s_a[NPROT];
    __shared__ float s_red[32][NPROT];

    const int cnt = counts[k];
    const float invBk = 1.f / (float)((cnt > 0) ? cnt : 1);
    const int* lst = lists + (size_t)k * NPIX;

    for (int it = 0; it < 3; it++) {
        float acc[NPROT];
        for (int m = 0; m < NPROT; m++) acc[m] = 0.f;
        for (int idx = tid; idx < cnt; idx += 1024) {
            const int n = lst[idx];
            const float bj = (it == 0) ? 1.f : bbuf[n];
            const float* pr = pl + (size_t)n * NJ + k * NPROT;
            for (int m = 0; m < NPROT; m++) acc[m] += __expf(pr[m] * 20.f) * bj;
        }
        for (int m = 0; m < NPROT; m++) {
            float v = waveRedSum(acc[m]);
            if (lane == 0) s_red[wv][m] = v;
        }
        __syncthreads();
        if (tid < NPROT) {
            float s = 0.f;
            for (int w = 0; w < 32; w++) s += s_red[w][tid];
            s_a[tid] = 1.f / (10.f * fmaxf(s, 1e-12f));
        }
        __syncthreads();
        for (int idx = tid; idx < cnt; idx += 1024) {
            const int n = lst[idx];
            const float* pr = pl + (size_t)n * NJ + k * NPROT;
            float col = 0.f;
            for (int m = 0; m < NPROT; m++) col += __expf(pr[m] * 20.f) * s_a[m];
            bbuf[n] = invBk / fmaxf(col, 1e-12f);
        }
        __syncthreads();
    }
    // argmax over m of E*a (b uniform in m): use log domain
    for (int idx = tid; idx < cnt; idx += 1024) {
        const int n = lst[idx];
        const float* pr = pl + (size_t)n * NJ + k * NPROT;
        int bm = 0; float bv = -1e30f;
        for (int m = 0; m < NPROT; m++) {
            const float v = pr[m] * 20.f + __logf(s_a[m]);
            if (v > bv) { bv = v; bm = m; }
        }
        picked[n] = bm;
        outPT[n] = (float)(bm + NPROT * k);
    }
}

// ---------------- K6: per-class f accumulation in LDS ----------------
__global__ void protoAccumKernel(const unsigned short* __restrict__ wsC,
                                 const int* __restrict__ counts,
                                 const int* __restrict__ lists,
                                 const int* __restrict__ correct,
                                 const int* __restrict__ picked,
                                 float* __restrict__ fglob,
                                 int* __restrict__ ncnt) {
    const int k = blockIdx.x;
    __shared__ float fl[NPROT][DIMD];
    __shared__ int   cl[NPROT];
    for (int i = threadIdx.x; i < NPROT * DIMD; i += 256) ((float*)fl)[i] = 0.f;
    if (threadIdx.x < NPROT) cl[threadIdx.x] = 0;
    __syncthreads();

    const int cnt = counts[k];
    const int* lst = lists + (size_t)k * NPIX;
    for (int p = 0; p < cnt; p++) {
        const int n = lst[p];
        if (!correct[n]) continue;                     // uniform across block
        const int m = picked[n];
        const unsigned short* rowc = wsC + (size_t)n * DIMP;
        for (int d = threadIdx.x; d < DIMD; d += 256) fl[m][d] += bf2f(rowc[d]);
        if (threadIdx.x == 0) cl[m]++;
    }
    __syncthreads();
    for (int i = threadIdx.x; i < NPROT * DIMD; i += 256)
        fglob[(size_t)k * NPROT * DIMD + i] = ((float*)fl)[i];
    if (threadIdx.x < NPROT) ncnt[k * NPROT + threadIdx.x] = cl[threadIdx.x];
}

// ---------------- K7: EMA update + l2 renorm -> protos_new ----------------
__global__ void protoUpdateKernel(const float* __restrict__ protos,
                                  const float* __restrict__ fglob,
                                  const int* __restrict__ ncnt,
                                  float* __restrict__ outPN) {
    const int km = blockIdx.x;                         // 0..189
    __shared__ float red[8];
    __shared__ float rowNew[DIMD];
    const float* po = protos + (size_t)km * DIMD;
    const float* fr = fglob + (size_t)km * DIMD;

    float ssp = 0.f, ssf = 0.f;
    for (int d = threadIdx.x; d < DIMD; d += 256) {
        const float a = po[d]; ssp += a * a;
        const float b = fr[d]; ssf += b * b;
    }
    const float sp = blockRedSum256(ssp, red);
    const float sf = blockRedSum256(ssf, red);
    const float rip = 1.f / fmaxf(sqrtf(sp), 1e-12f);
    const float rif = 1.f / fmaxf(sqrtf(sf), 1e-12f);
    const bool upd = ncnt[km] > 0;

    float ssn = 0.f;
    for (int d = threadIdx.x; d < DIMD; d += 256) {
        const float pn = po[d] * rip;
        const float v = upd ? (0.999f * pn + 0.001f * (fr[d] * rif)) : pn;
        rowNew[d] = v; ssn += v * v;
    }
    const float sn = blockRedSum256(ssn, red);
    const float rin = 1.f / fmaxf(sqrtf(sn), 1e-12f);
    for (int d = threadIdx.x; d < DIMD; d += 256)
        outPN[(size_t)km * DIMD + d] = rowNew[d] * rin;
}

// ---------------- host launch ----------------
extern "C" void kernel_launch(void* const* d_in, const int* in_sizes, int n_in,
                              void* d_out, int out_size, void* d_ws, size_t ws_size,
                              hipStream_t stream) {
    const float* c      = (const float*)d_in[0];
    const int*   gt     = (const int*)  d_in[1];
    const float* protos = (const float*)d_in[2];
    const float* fg     = (const float*)d_in[3];
    const float* fb     = (const float*)d_in[4];
    const float* mg     = (const float*)d_in[5];
    const float* mb     = (const float*)d_in[6];

    // output layout: out_seg | proto_logits | proto_target | protos_new
    float* outSeg = (float*)d_out;                                  // 4*19*128*256
    float* outPL  = outSeg + (size_t)4 * NCLS * HW;                 // NPIX*190
    float* outPT  = outPL + (size_t)NPIX * NJ;                      // NPIX
    float* outPN  = outPT + (size_t)NPIX;                           // 190*720

    // workspace layout
    char* ws = (char*)d_ws;
    size_t off = 0;
    unsigned short* wsC = (unsigned short*)(ws + off); off += (size_t)NPIX * DIMP * 2;
    unsigned short* wsP = (unsigned short*)(ws + off); off += (size_t)192 * DIMP * 2;
    int*   picked  = (int*)  (ws + off); off += (size_t)NPIX * 4;
    int*   correct = (int*)  (ws + off); off += (size_t)NPIX * 4;
    int*   counts  = (int*)  (ws + off); off += 256;
    int*   lists   = (int*)  (ws + off); off += (size_t)NCLS * NPIX * 4;
    float* bbuf    = (float*)(ws + off); off += (size_t)NPIX * 4;
    float* fglob   = (float*)(ws + off); off += (size_t)NJ * DIMD * 4;
    int*   ncnt    = (int*)  (ws + off); off += (size_t)NJ * 4;
    (void)ws_size; (void)in_sizes; (void)n_in; (void)out_size;

    (void)hipMemsetAsync(counts, 0, 256, stream);

    featNormKernel  <<<NPIX / 32, 256, 0, stream>>>(c, fg, fb, wsC);
    protoNormKernel <<<192, 256, 0, stream>>>(protos, wsP);
    gemmKernel      <<<NPIX / 128, 256, 0, stream>>>(wsC, wsP, outPL);
    segPostKernel   <<<NPIX / 8, 256, 0, stream>>>(outPL, gt, mg, mb, outSeg, correct, counts, lists);
    sinkhornKernel  <<<NCLS, 1024, 0, stream>>>(outPL, counts, lists, bbuf, picked, outPT);
    protoAccumKernel<<<NCLS, 256, 0, stream>>>(wsC, counts, lists, correct, picked, fglob, ncnt);
    protoUpdateKernel<<<NJ, 256, 0, stream>>>(protos, fglob, ncnt, outPN);
}
